// NGP_1726576857412
// MI455X (gfx1250) — matrix operand order repaired and checked by reading, not compiled, not run on hardware
//
#include <hip/hip_runtime.h>
#include <stdint.h>
#include <math.h>

// ---------------------------------------------------------------------------
// Instant-NGP NeRF pipeline for MI455X (gfx1250, wave32, WMMA).
//
// Roofline: hash-table gathers (192 MB tables ~= global L2) dominate bytes;
// MLP heads dominate FLOPs and are batched into v_wmma_f32_16x16x32_bf16
// (16 points x 16 outs x 32 K per instruction, f32 accumulate).
// ---------------------------------------------------------------------------

#define N_PTS   262144
#define T_TAB   (1u << 19)
#define TMASK   (T_TAB - 1u)
#define HP1     2654435761u
#define HP2     805459861u

typedef __attribute__((ext_vector_type(16))) __bf16 v16bf;
typedef __attribute__((ext_vector_type(8)))  float  v8f;

// floor(16 * 2^(7l/15)) and floor(16 * 2^(7l/31)) (matches numpy float64 path)
__constant__ float RESX[16] = {16.f,22.f,30.f,42.f,58.f,80.f,111.f,153.f,
                               212.f,294.f,406.f,561.f,776.f,1072.f,1482.f,2048.f};
__constant__ float RESR[32] = {16.f,18.f,21.f,25.f,29.f,34.f,40.f,47.f,55.f,65.f,
                               76.f,89.f,104.f,122.f,143.f,167.f,195.f,228.f,267.f,
                               313.f,366.f,428.f,500.f,585.f,684.f,800.f,936.f,
                               1095.f,1280.f,1497.f,1751.f,2048.f};

// packed-weight workspace offsets (uint32 words; each word = 2 bf16)
enum {
  OFF_NW1 = 0,      // norm_w1 64x64  : 2 x 4 tiles
  OFF_NW2 = 2048,   // norm_w2 64x64
  OFF_NW3 = 4096,   // norm_w3 64x3 (N pad 16) : 2 x 1 tiles
  OFF_SW1 = 4608,
  OFF_SW2 = 6656,
  OFF_SW3 = 8704,
  OFF_RW1 = 9216,   // rgb_w1 80x128 (K pad 96): 3 x 8 tiles
  OFF_RW2 = 15360,  // rgb_w2 128x128: 4 x 8 tiles
  OFF_RW3 = 23552,  // rgb_w3 128x3 (N pad 16): 4 x 1 tiles
  OFF_FEAT = 24576  // bf16 feat_rgb records: N_PTS * 32 words
};

// ----------------------------- helpers -------------------------------------

__device__ __forceinline__ uint32_t bf16_rne(float f) {
  uint32_t u = __float_as_uint(f);
  return ((u + 0x7FFFu + ((u >> 16) & 1u)) >> 16) & 0xFFFFu;
}
__device__ __forceinline__ uint32_t pk_bf16(float a, float b) {
  return bf16_rne(a) | (bf16_rne(b) << 16);
}

union AFrag { v16bf v; uint32_t w[8]; };
union BFrag { v16bf v; uint32_t w[8]; };
union CFrag { v8f  v; float    f[8]; };

// A fragment (16x32 bf16) from an LDS record of bf16-pair words.
// ISA layout: lane l holds row M=l%16; VGPR v holds K pair
// k0 = (v<4 ? 2v : 16+2(v-4)) + 8*(l/16). word index = word0 + k0/2.
__device__ __forceinline__ v16bf load_a_words(const uint32_t* rec, int stride,
                                              int word0, int lane) {
  int m = lane & 15, h = lane >> 4;
  AFrag a;
#pragma unroll
  for (int v = 0; v < 8; ++v) {
    int k0 = (v < 4 ? 2 * v : 16 + 2 * (v - 4)) + 8 * h;
    a.w[v] = rec[m * stride + word0 + (k0 >> 1)];
  }
  return a.v;
}

// A fragment from an f32 LDS activation buffer (convert + pack to bf16).
__device__ __forceinline__ v16bf load_a_f32(const float* hbuf, int stride,
                                            int kbase, int lane) {
  int m = lane & 15, h = lane >> 4;
  AFrag a;
#pragma unroll
  for (int v = 0; v < 8; ++v) {
    int k0 = kbase + (v < 4 ? 2 * v : 16 + 2 * (v - 4)) + 8 * h;
    a.w[v] = pk_bf16(hbuf[m * stride + k0], hbuf[m * stride + k0 + 1]);
  }
  return a.v;
}

// B fragment (32x16 bf16) from pre-packed global tiles: lane-contiguous,
// two global_load_b128 per lane.
__device__ __forceinline__ v16bf load_b(const uint32_t* base, int tile, int lane) {
  const uint4* p = (const uint4*)(base + ((size_t)tile << 8) + lane * 8);
  uint4 q0 = p[0], q1 = p[1];
  BFrag b;
  b.w[0] = q0.x; b.w[1] = q0.y; b.w[2] = q0.z; b.w[3] = q0.w;
  b.w[4] = q1.x; b.w[5] = q1.y; b.w[6] = q1.z; b.w[7] = q1.w;
  return b.v;
}

// C store with fused bias + relu into f32 LDS activation buffer.
// C layout: VGPR r, lane l -> row m = r + 8*(l/16), col n = l%16.
__device__ __forceinline__ void store_c_relu(float* hbuf, int stride, int nt,
                                             const float* bias, const CFrag& c,
                                             int lane) {
  int n = nt * 16 + (lane & 15);
  int hg = lane >> 4;
  float bv = bias[n];
#pragma unroll
  for (int r = 0; r < 8; ++r) {
    float val = c.f[r] + bv;
    hbuf[(r + 8 * hg) * stride + n] = val > 0.f ? val : 0.f;
  }
}

#define WMMA_BF16(A, B, C) \
  __builtin_amdgcn_wmma_f32_16x16x32_bf16(false, (A), false, (B), (short)0, (C), false, false)

__device__ __forceinline__ void sh_encode(float x, float y, float z, float* o) {
  float x2 = x * x, y2 = y * y, z2 = z * z;
  float xy = x * y, yz = y * z, xz = x * z;
  o[0]  = 0.28209479177387814f;
  o[1]  = -0.48860251190291987f * y;
  o[2]  =  0.48860251190291987f * z;
  o[3]  = -0.48860251190291987f * x;
  o[4]  =  1.0925484305920792f * xy;
  o[5]  = -1.0925484305920792f * yz;
  o[6]  =  0.94617469575756f * z2 - 0.31539156525252005f;
  o[7]  = -1.0925484305920792f * xz;
  o[8]  =  0.5462742152960396f * (x2 - y2);
  o[9]  = -0.5900435899266435f * y * (3.f * x2 - y2);
  o[10] =  2.890611442640554f * xy * z;
  o[11] =  0.4570457994644657f * y * (1.f - 5.f * z2);
  o[12] =  0.3731763325901154f * z * (5.f * z2 - 3.f);
  o[13] =  0.4570457994644657f * x * (1.f - 5.f * z2);
  o[14] =  1.445305721320277f * z * (x2 - y2);
  o[15] = -0.5900435899266435f * x * (x2 - 3.f * y2);
}

// ----------------------------- kernels -------------------------------------

__global__ void ngp_zero_kernel(float* p) {
  if (threadIdx.x == 0 && blockIdx.x == 0) *p = 0.f;
}

// Repack a (K x Nw) row-major f32 weight into bf16 WMMA B-fragment tiles.
__global__ void ngp_pack_kernel(const float* __restrict__ W, int K, int Nw,
                                int NT, uint32_t* __restrict__ out, int nwords) {
  int w = blockIdx.x * blockDim.x + threadIdx.x;
  if (w >= nwords) return;
  int tile = w >> 8, r = w & 255, lane = r >> 3, v = r & 7;
  int h = lane >> 4, nl = lane & 15;
  int kt = tile / NT, nt = tile % NT;
  int k = kt * 32 + (v < 4 ? 2 * v : 16 + 2 * (v - 4)) + 8 * h;
  int n = nt * 16 + nl;
  float f0 = (k < K && n < Nw) ? W[k * Nw + n] : 0.f;
  float f1 = (k + 1 < K && n < Nw) ? W[(k + 1) * Nw + n] : 0.f;
  out[w] = pk_bf16(f0, f1);
}

// rgb hash-grid encode -> bf16-packed feature records (32 words/point).
__global__ __launch_bounds__(256) void ngp_featrgb_kernel(
    const float* __restrict__ x, const float* __restrict__ tab,
    uint32_t* __restrict__ featout) {
  int i = blockIdx.x * blockDim.x + threadIdx.x;
  if (i >= N_PTS) return;
  float xn0 = (x[i * 3 + 0] + 1.f) * 0.5f;
  float xn1 = (x[i * 3 + 1] + 1.f) * 0.5f;
  float xn2 = (x[i * 3 + 2] + 1.f) * 0.5f;
#pragma unroll
  for (int l = 0; l < 32; ++l) {
    float res = RESR[l];
    float px = xn0 * res, py = xn1 * res, pz = xn2 * res;
    float fx = floorf(px), fy = floorf(py), fz = floorf(pz);
    float wx = px - fx, wy = py - fy, wz = pz - fz;
    uint32_t cx = (uint32_t)fx, cy = (uint32_t)fy, cz = (uint32_t)fz;
    const float* tb = tab + (size_t)l * T_TAB * 2;
    float f0 = 0.f, f1 = 0.f;
#pragma unroll
    for (int c = 0; c < 8; ++c) {
      int dx = c & 1, dy = (c >> 1) & 1, dz = (c >> 2) & 1;
      uint32_t idx = ((cx + dx) ^ ((cy + dy) * HP1) ^ ((cz + dz) * HP2)) & TMASK;
      float2 t = *(const float2*)(tb + (size_t)idx * 2);
      float ax = dx ? wx : 1.f - wx;
      float ay = dy ? wy : 1.f - wy;
      float az = dz ? wz : 1.f - wz;
      float www = ax * ay * az;
      f0 += www * t.x;
      f1 += www * t.y;
    }
    featout[(size_t)i * 32 + l] = pk_bf16(f0, f1);
  }
}

// xyz branch: hash encode + analytic VJP + tiny MLP fwd/bwd.
__global__ __launch_bounds__(256) void ngp_sigma_kernel(
    const float* __restrict__ x, const float* __restrict__ tab,
    const float* __restrict__ w1, const float* __restrict__ b1,
    const float* __restrict__ w2, const float* __restrict__ b2,
    float* __restrict__ sigmas, float* __restrict__ nraw,
    float* __restrict__ cnt) {
  int i = blockIdx.x * blockDim.x + threadIdx.x;
  if (i >= N_PTS) return;
  float xn0 = (x[i * 3 + 0] + 1.f) * 0.5f;
  float xn1 = (x[i * 3 + 1] + 1.f) * 0.5f;
  float xn2 = (x[i * 3 + 2] + 1.f) * 0.5f;
  float h[32], dfx[32], dfy[32], dfz[32];
#pragma unroll
  for (int l = 0; l < 16; ++l) {
    float res = RESX[l];
    float px = xn0 * res, py = xn1 * res, pz = xn2 * res;
    float fx = floorf(px), fy = floorf(py), fz = floorf(pz);
    float wx = px - fx, wy = py - fy, wz = pz - fz;
    uint32_t cx = (uint32_t)fx, cy = (uint32_t)fy, cz = (uint32_t)fz;
    const float* tb = tab + (size_t)l * T_TAB * 2;
    float f0 = 0.f, f1 = 0.f;
    float gx0 = 0.f, gx1 = 0.f, gy0 = 0.f, gy1 = 0.f, gz0 = 0.f, gz1 = 0.f;
#pragma unroll
    for (int c = 0; c < 8; ++c) {
      int dx = c & 1, dy = (c >> 1) & 1, dz = (c >> 2) & 1;
      uint32_t idx = ((cx + dx) ^ ((cy + dy) * HP1) ^ ((cz + dz) * HP2)) & TMASK;
      float2 t = *(const float2*)(tb + (size_t)idx * 2);
      float ax = dx ? wx : 1.f - wx, sx = dx ? 1.f : -1.f;
      float ay = dy ? wy : 1.f - wy, sy = dy ? 1.f : -1.f;
      float az = dz ? wz : 1.f - wz, sz = dz ? 1.f : -1.f;
      float www = ax * ay * az;
      f0 += www * t.x; f1 += www * t.y;
      float gx = sx * ay * az, gy = ax * sy * az, gz = ax * ay * sz;
      gx0 += gx * t.x; gx1 += gx * t.y;
      gy0 += gy * t.x; gy1 += gy * t.y;
      gz0 += gz * t.x; gz1 += gz * t.y;
    }
    float sc = res * 0.5f;  // d(pos)/d(x) = res * dxn/dx, dxn/dx = 1/(2*SCALE)
    h[2 * l] = f0;        h[2 * l + 1] = f1;
    dfx[2 * l] = gx0 * sc; dfx[2 * l + 1] = gx1 * sc;
    dfy[2 * l] = gy0 * sc; dfy[2 * l + 1] = gy1 * sc;
    dfz[2 * l] = gz0 * sc; dfz[2 * l + 1] = gz1 * sc;
  }
  // forward 32 -> 64 -> 1 (weight addresses thread-uniform -> scalar loads)
  float g = b2[0];
  uint64_t mask = 0ull;
  for (int j = 0; j < 64; ++j) {
    float z = b1[j];
#pragma unroll
    for (int f = 0; f < 32; ++f) z = fmaf(h[f], w1[f * 64 + j], z);
    if (z > 0.f) { g = fmaf(z, w2[j], g); mask |= (1ull << j); }
  }
  float t1 = g - 1.f;
  sigmas[i] = (t1 > 20.f) ? t1 : log1pf(__expf(t1));
  float sig = 1.f / (1.f + __expf(-t1));  // softplus'
  // backward: grad_x[a] = sig * sum_j relu'_j W2_j * sum_f W1[f,j] * dfeat_f/dx_a
  float gx = 0.f, gy = 0.f, gz = 0.f;
  for (int j = 0; j < 64; ++j) {
    if (!((mask >> j) & 1ull)) continue;
    float wj = w2[j];
#pragma unroll
    for (int f = 0; f < 32; ++f) {
      float coef = w1[f * 64 + j] * wj;
      gx = fmaf(coef, dfx[f], gx);
      gy = fmaf(coef, dfy[f], gy);
      gz = fmaf(coef, dfz[f], gz);
    }
  }
  gx *= sig; gy *= sig; gz *= sig;
  int c = (isinf(gx) ? 1 : 0) + (isinf(gy) ? 1 : 0) + (isinf(gz) ? 1 : 0);
  if (c) atomicAdd(cnt, (float)c);
  float nrm = sqrtf(gx * gx + gy * gy + gz * gz);
  float inv = -1.f / fmaxf(nrm, 1e-6f);
  nraw[i * 3 + 0] = gx * inv;
  nraw[i * 3 + 1] = gy * inv;
  nraw[i * 3 + 2] = gz * inv;
}

// Shared 64-wide head (norm / sem): 64 ->relu-> 64 ->relu-> 64 -> nout.
// All 4 waves run layer 3 redundantly (same tile, identical LDS writes) so
// every WMMA executes with EXEC all-ones (ISA 7.12 restriction).
__device__ __forceinline__ void run_head64(
    const uint32_t* s_in, float* s_h1, float* s_h2,
    const uint32_t* wpack, int ow1, int ow2, int ow3,
    const float* b1, const float* b2, const float* b3, int nout,
    int lane, int wv) {
  CFrag c;
#pragma unroll
  for (int r = 0; r < 8; ++r) c.f[r] = 0.f;
#pragma unroll
  for (int kt = 0; kt < 2; ++kt) {
    v16bf a = load_a_words(s_in, 48, 8 + kt * 16, lane);  // feat at words 8..39
    v16bf b = load_b(wpack + ow1, kt * 4 + wv, lane);
    c.v = WMMA_BF16(a, b, c.v);
  }
  store_c_relu(s_h1, 64, wv, b1, c, lane);
  __syncthreads();
#pragma unroll
  for (int r = 0; r < 8; ++r) c.f[r] = 0.f;
#pragma unroll
  for (int kt = 0; kt < 2; ++kt) {
    v16bf a = load_a_f32(s_h1, 64, kt * 32, lane);
    v16bf b = load_b(wpack + ow2, kt * 4 + wv, lane);
    c.v = WMMA_BF16(a, b, c.v);
  }
  store_c_relu(s_h2, 64, wv, b2, c, lane);
  __syncthreads();
#pragma unroll
  for (int r = 0; r < 8; ++r) c.f[r] = 0.f;
#pragma unroll
  for (int kt = 0; kt < 2; ++kt) {
    v16bf a = load_a_f32(s_h2, 64, kt * 32, lane);
    v16bf b = load_b(wpack + ow3, kt, lane);
    c.v = WMMA_BF16(a, b, c.v);
  }
  {
    int n = lane & 15, hg = lane >> 4;
    float bv = (n < nout) ? b3[n] : 0.f;
#pragma unroll
    for (int r = 0; r < 8; ++r) s_h1[(r + 8 * hg) * 16 + n] = c.f[r] + bv;
  }
  __syncthreads();
}

__global__ __launch_bounds__(128) void ngp_heads_kernel(
    const uint32_t* __restrict__ wpack, const uint32_t* __restrict__ feat,
    const float* __restrict__ dirs,
    const float* __restrict__ nb1, const float* __restrict__ nb2, const float* __restrict__ nb3,
    const float* __restrict__ sb1, const float* __restrict__ sb2, const float* __restrict__ sb3,
    const float* __restrict__ rb1, const float* __restrict__ rb2, const float* __restrict__ rb3,
    float* __restrict__ out_rgb, float* __restrict__ out_np,
    float* __restrict__ out_sem) {
  __shared__ uint32_t s_in[16 * 48];   // per point: sh[8w] | feat[32w] | pad[8w]
  __shared__ float    s_h1[16 * 128];
  __shared__ float    s_h2[16 * 128];
  const int pt0 = blockIdx.x * 16;
  const int tid = threadIdx.x, lane = tid & 31, wv = tid >> 5;

  // stage feature tile (512 words) into LDS
  for (int k = tid; k < 16 * 32; k += 128) {
    int p = k >> 5, wd = k & 31;
    s_in[p * 48 + 8 + wd] = feat[(size_t)(pt0 + p) * 32 + wd];
  }
  { // zero K-pad (K 80..95 of the rgb-head input)
    int p = tid >> 3, wd = tid & 7;
    s_in[p * 48 + 40 + wd] = 0u;
  }
  if (tid < 16) {  // SH-deg4 of normalized direction, bf16-packed words 0..7
    float dx = dirs[(pt0 + tid) * 3 + 0];
    float dy = dirs[(pt0 + tid) * 3 + 1];
    float dz = dirs[(pt0 + tid) * 3 + 2];
    float n = sqrtf(dx * dx + dy * dy + dz * dz);
    float inv = 1.f / fmaxf(n, 1e-6f);
    dx *= inv; dy *= inv; dz *= inv;
    float s[16];
    sh_encode(dx, dy, dz, s);
#pragma unroll
    for (int c2 = 0; c2 < 8; ++c2) s_in[tid * 48 + c2] = pk_bf16(s[2 * c2], s[2 * c2 + 1]);
  }
  __syncthreads();

  // ---------------- norm head: normals_pred = -l2norm(mlp(feat)) ----------
  run_head64(s_in, s_h1, s_h2, wpack, OFF_NW1, OFF_NW2, OFF_NW3, nb1, nb2, nb3, 3, lane, wv);
  if (tid < 16) {
    float a0 = s_h1[tid * 16 + 0], a1 = s_h1[tid * 16 + 1], a2 = s_h1[tid * 16 + 2];
    float nn = sqrtf(a0 * a0 + a1 * a1 + a2 * a2);
    float inv = -1.f / fmaxf(nn, 1e-6f);
    out_np[(size_t)(pt0 + tid) * 3 + 0] = a0 * inv;
    out_np[(size_t)(pt0 + tid) * 3 + 1] = a1 * inv;
    out_np[(size_t)(pt0 + tid) * 3 + 2] = a2 * inv;
  }
  __syncthreads();

  // ---------------- sem head: softmax7(mlp(feat)) -------------------------
  run_head64(s_in, s_h1, s_h2, wpack, OFF_SW1, OFF_SW2, OFF_SW3, sb1, sb2, sb3, 7, lane, wv);
  if (tid < 16) {
    float v[7], m = -1e30f;
#pragma unroll
    for (int j = 0; j < 7; ++j) { v[j] = s_h1[tid * 16 + j]; m = fmaxf(m, v[j]); }
    float s = 0.f;
#pragma unroll
    for (int j = 0; j < 7; ++j) { v[j] = __expf(v[j] - m); s += v[j]; }
    float inv = 1.f / s;
#pragma unroll
    for (int j = 0; j < 7; ++j) out_sem[(size_t)(pt0 + tid) * 7 + j] = v[j] * inv;
  }
  __syncthreads();

  // ---------------- rgb head: [sh|feat](96) -> 128 -> 128 -> 3 ------------
  {
    CFrag c0, c1;
#pragma unroll
    for (int r = 0; r < 8; ++r) { c0.f[r] = 0.f; c1.f[r] = 0.f; }
#pragma unroll
    for (int kt = 0; kt < 3; ++kt) {  // K = 96 (80 real + 16 zero pad)
      v16bf a = load_a_words(s_in, 48, kt * 16, lane);
      v16bf b0 = load_b(wpack + OFF_RW1, kt * 8 + wv, lane);
      c0.v = WMMA_BF16(a, b0, c0.v);
      v16bf b1f = load_b(wpack + OFF_RW1, kt * 8 + wv + 4, lane);
      c1.v = WMMA_BF16(a, b1f, c1.v);
    }
    store_c_relu(s_h1, 128, wv, rb1, c0, lane);
    store_c_relu(s_h1, 128, wv + 4, rb1, c1, lane);
    __syncthreads();
#pragma unroll
    for (int r = 0; r < 8; ++r) { c0.f[r] = 0.f; c1.f[r] = 0.f; }
#pragma unroll
    for (int kt = 0; kt < 4; ++kt) {  // K = 128
      v16bf a = load_a_f32(s_h1, 128, kt * 32, lane);
      v16bf b0 = load_b(wpack + OFF_RW2, kt * 8 + wv, lane);
      c0.v = WMMA_BF16(a, b0, c0.v);
      v16bf b1f = load_b(wpack + OFF_RW2, kt * 8 + wv + 4, lane);
      c1.v = WMMA_BF16(a, b1f, c1.v);
    }
    store_c_relu(s_h2, 128, wv, rb2, c0, lane);
    store_c_relu(s_h2, 128, wv + 4, rb2, c1, lane);
    __syncthreads();
#pragma unroll
    for (int r = 0; r < 8; ++r) c0.f[r] = 0.f;
#pragma unroll
    for (int kt = 0; kt < 4; ++kt) {  // layer 3, all waves redundantly
      v16bf a = load_a_f32(s_h2, 128, kt * 32, lane);
      v16bf b0 = load_b(wpack + OFF_RW3, kt, lane);
      c0.v = WMMA_BF16(a, b0, c0.v);
    }
    {
      int n = lane & 15, hg = lane >> 4;
      float bv = (n < 3) ? rb3[n] : 0.f;
#pragma unroll
      for (int r = 0; r < 8; ++r) s_h1[(r + 8 * hg) * 16 + n] = c0.f[r] + bv;
    }
    __syncthreads();
    if (tid < 16) {
#pragma unroll
      for (int j = 0; j < 3; ++j) {
        float v = s_h1[tid * 16 + j];
        out_rgb[(size_t)(pt0 + tid) * 3 + j] = 1.f / (1.f + __expf(-v));
      }
    }
  }
}

// ----------------------------- launch --------------------------------------

extern "C" void kernel_launch(void* const* d_in, const int* in_sizes, int n_in,
                              void* d_out, int out_size, void* d_ws, size_t ws_size,
                              hipStream_t stream) {
  (void)in_sizes; (void)n_in; (void)out_size; (void)ws_size;
  const float* x        = (const float*)d_in[0];
  const float* d        = (const float*)d_in[1];
  const float* xyz_tab  = (const float*)d_in[2];
  const float* rgb_tab  = (const float*)d_in[3];
  const float* xyz_w1   = (const float*)d_in[4];
  const float* xyz_b1   = (const float*)d_in[5];
  const float* xyz_w2   = (const float*)d_in[6];
  const float* xyz_b2   = (const float*)d_in[7];
  const float* norm_w1  = (const float*)d_in[8];
  const float* norm_b1  = (const float*)d_in[9];
  const float* norm_w2  = (const float*)d_in[10];
  const float* norm_b2  = (const float*)d_in[11];
  const float* norm_w3  = (const float*)d_in[12];
  const float* norm_b3  = (const float*)d_in[13];
  const float* sem_w1   = (const float*)d_in[14];
  const float* sem_b1   = (const float*)d_in[15];
  const float* sem_w2   = (const float*)d_in[16];
  const float* sem_b2   = (const float*)d_in[17];
  const float* sem_w3   = (const float*)d_in[18];
  const float* sem_b3   = (const float*)d_in[19];
  const float* rgb_w1   = (const float*)d_in[20];
  const float* rgb_b1   = (const float*)d_in[21];
  const float* rgb_w2   = (const float*)d_in[22];
  const float* rgb_b2   = (const float*)d_in[23];
  const float* rgb_w3   = (const float*)d_in[24];
  const float* rgb_b3   = (const float*)d_in[25];

  float* out      = (float*)d_out;
  float* o_sig    = out;                       // N
  float* o_rgb    = o_sig   + N_PTS;           // 3N
  float* o_nraw   = o_rgb   + 3 * (size_t)N_PTS;
  float* o_npred  = o_nraw  + 3 * (size_t)N_PTS;
  float* o_sem    = o_npred + 3 * (size_t)N_PTS;
  float* o_cnt    = o_sem   + 7 * (size_t)N_PTS;  // 1

  uint32_t* wpack   = (uint32_t*)d_ws;            // 24576 words (96 KB)
  uint32_t* featbuf = wpack + OFF_FEAT;           // N_PTS*32 words (32 MB)

  ngp_zero_kernel<<<1, 32, 0, stream>>>(o_cnt);

  auto pack = [&](const float* W, int K, int Nw, int NT, int off, int nwords) {
    ngp_pack_kernel<<<(nwords + 255) / 256, 256, 0, stream>>>(W, K, Nw, NT,
                                                              wpack + off, nwords);
  };
  pack(norm_w1, 64, 64, 4, OFF_NW1, 2048);
  pack(norm_w2, 64, 64, 4, OFF_NW2, 2048);
  pack(norm_w3, 64, 3, 1, OFF_NW3, 512);
  pack(sem_w1, 64, 64, 4, OFF_SW1, 2048);
  pack(sem_w2, 64, 64, 4, OFF_SW2, 2048);
  pack(sem_w3, 64, 7, 1, OFF_SW3, 512);
  pack(rgb_w1, 80, 128, 8, OFF_RW1, 6144);
  pack(rgb_w2, 128, 128, 8, OFF_RW2, 8192);
  pack(rgb_w3, 128, 3, 1, OFF_RW3, 1024);

  ngp_featrgb_kernel<<<N_PTS / 256, 256, 0, stream>>>(x, rgb_tab, featbuf);
  ngp_sigma_kernel<<<N_PTS / 256, 256, 0, stream>>>(x, xyz_tab, xyz_w1, xyz_b1,
                                                    xyz_w2, xyz_b2, o_sig, o_nraw,
                                                    o_cnt);
  ngp_heads_kernel<<<N_PTS / 16, 128, 0, stream>>>(
      wpack, featbuf, d, norm_b1, norm_b2, norm_b3, sem_b1, sem_b2, sem_b3,
      rgb_b1, rgb_b2, rgb_b3, o_rgb, o_npred, o_sem);
}
